// VectorQuantizer_19327352832254
// MI455X (gfx1250) — matrix operand order repaired and checked by reading, not compiled, not run on hardware
//
#include <hip/hip_runtime.h>

// ---------------------------------------------------------------------------
// VQ nearest-codebook search for MI455X (gfx1250, wave32, WMMA).
//   dists = ||z||^2 - 2 z.W^T + ||w||^2 ; argmin_k ; gather ; ST estimator.
// ||z||^2 is row-constant -> dropped from the argmin.
// Score matmul runs on v_wmma_f32_16x16x32_bf16 with a hi/lo bf16 split:
//   z ~ zh+zl, w ~ wh+wl, score ~ zh*wh + zh*wl + zl*wh  (fp32-class accuracy)
// Weight tiles are double-buffered in LDS via global_load_async_to_lds_b128
// (ASYNCcnt pipeline) so the 8 waves never stall on tile staging.
// ---------------------------------------------------------------------------

typedef __attribute__((ext_vector_type(16))) __bf16 v16bf;
typedef __attribute__((ext_vector_type(8)))  float  v8f;
typedef __attribute__((ext_vector_type(4)))  float  f32x4;

#define B_ROWS   65536
#define K_CODES  1024
#define DIM      256
#define ROWS_PER_BLOCK 128          // 8 waves * 16 rows
#define NTILES   (K_CODES / 16)     // 64
#define KSTEPS   (DIM / 32)         // 8
#define TILE_BYTES 8192             // 16 codes * 256 dims * 2B (per hi or lo)

// LDS layout (bytes)
#define OFF_ZHI   0                         // 128*256*2 = 65536
#define OFF_ZLO   65536                     // 65536
#define OFF_B     131072                    // 2 buffers * (hi 8192 + lo 8192) = 32768
#define OFF_WN    163840                    // 1024*4    = 4096
#define OFF_IDX   167936                    // 128*4     = 512
#define OFF_RED   168448                    // 8*4       = 32
#define SMEM_BYTES 168480

// ---------------------------------------------------------------------------
// Async copy of one 16-code tile (hi + lo planes) into LDS.
// Per wave: 4 x global_load_async_to_lds_b128 (ASYNCcnt += 4).
// GVS addressing: vdst = LDS byte address, vaddr = 32-bit byte offset,
// saddr = 64-bit SGPR base.
// ---------------------------------------------------------------------------
__device__ __forceinline__ void async_tile16(unsigned ldsBase,
                                             const void* hiSrc, const void* loSrc,
                                             int tid)
{
    const unsigned o0 = (unsigned)tid * 16u;      // 256 threads x 16B = 4KB
    const unsigned o1 = o0 + 4096u;               // second 4KB half
    asm volatile(
        "global_load_async_to_lds_b128 %0, %4, %6\n\t"
        "global_load_async_to_lds_b128 %1, %5, %6\n\t"
        "global_load_async_to_lds_b128 %2, %4, %7\n\t"
        "global_load_async_to_lds_b128 %3, %5, %7"
        :: "v"(ldsBase + o0), "v"(ldsBase + o1),
           "v"(ldsBase + TILE_BYTES + o0), "v"(ldsBase + TILE_BYTES + o1),
           "v"(o0), "v"(o1),
           "s"(hiSrc), "s"(loSrc)
        : "memory");
}

// ---------------------------------------------------------------------------
// Prep: split codebook to bf16 hi/lo, compute ||w_k||^2.  Grid = K, block = D.
// ---------------------------------------------------------------------------
__global__ __launch_bounds__(DIM)
void vq_prep(const float* __restrict__ w,
             __bf16* __restrict__ whi, __bf16* __restrict__ wlo,
             float* __restrict__ wnorm)
{
    const int k = blockIdx.x;
    const int d = threadIdx.x;
    const float x = w[(size_t)k * DIM + d];
    const __bf16 h = (__bf16)x;
    whi[(size_t)k * DIM + d] = h;
    wlo[(size_t)k * DIM + d] = (__bf16)(x - (float)h);

    // deterministic wave + LDS reduction of x*x
    float s = x * x;
    #pragma unroll
    for (int m = 16; m >= 1; m >>= 1) s += __shfl_xor(s, m, 32);
    __shared__ float red[DIM / 32];
    if ((d & 31) == 0) red[d >> 5] = s;
    __syncthreads();
    if (d == 0) {
        float t = 0.f;
        #pragma unroll
        for (int i = 0; i < DIM / 32; ++i) t += red[i];
        wnorm[k] = t;
    }
}

// ---------------------------------------------------------------------------
// Main: per workgroup 128 z rows vs all 1024 codes.
// ---------------------------------------------------------------------------
__global__ __launch_bounds__(256)
void vq_main(const float* __restrict__ z, const float* __restrict__ w,
             const __bf16* __restrict__ whi, const __bf16* __restrict__ wlo,
             const float* __restrict__ wnorm,
             float* __restrict__ out, float* __restrict__ partials)
{
    extern __shared__ char smem[];
    __bf16* zhiS = (__bf16*)(smem + OFF_ZHI);
    __bf16* zloS = (__bf16*)(smem + OFF_ZLO);
    float*  wnS  = (float*)(smem + OFF_WN);
    int*    idxS = (int*)(smem + OFF_IDX);
    float*  redS = (float*)(smem + OFF_RED);

    const int tid  = threadIdx.x;
    const int lane = tid & 31;
    const int wave = tid >> 5;
    const int blockRow = blockIdx.x * ROWS_PER_BLOCK;

    // ---- kick off async staging of weight tile 0 into buffer 0 ----
    async_tile16(OFF_B, whi, wlo, tid);

    // ---- stage 128 z rows as bf16 hi/lo in LDS (vectorized fp32 reads) ----
    {
        const f32x4* z4 = (const f32x4*)(z + (size_t)blockRow * DIM);
        for (int i = tid; i < ROWS_PER_BLOCK * DIM / 4; i += 256) {
            f32x4 x = z4[i];
            #pragma unroll
            for (int c = 0; c < 4; ++c) {
                const float v = x[c];
                const __bf16 hh = (__bf16)v;
                zhiS[i * 4 + c] = hh;
                zloS[i * 4 + c] = (__bf16)(v - (float)hh);
            }
        }
        for (int i = tid; i < K_CODES; i += 256) wnS[i] = wnorm[i];
    }

    // per-lane running argmin. C-layout: VGPR r, lanes 0-15 -> row r,
    // lanes 16-31 -> row r+8; column N = lane%16.
    float bestD[8];
    int   bestI[8];
    #pragma unroll
    for (int r = 0; r < 8; ++r) { bestD[r] = 3.4e38f; bestI[r] = 0; }

    const int m = lane & 15;     // A: row / B,C: column
    const int h = lane >> 4;     // K-half selector
    const int rowL = wave * 16;  // this wave's local row base

    for (int nt = 0; nt < NTILES; ++nt) {
        const unsigned bufOff = OFF_B + (unsigned)(nt & 1) * (2 * TILE_BYTES);

        // prefetch next tile into the other buffer (safe: it was last read in
        // iteration nt-1, which ended with a barrier), then wait for tile nt.
        if (nt + 1 < NTILES) {
            const unsigned nxtOff = OFF_B + (unsigned)((nt + 1) & 1) * (2 * TILE_BYTES);
            async_tile16(nxtOff,
                         whi + (size_t)(nt + 1) * 16 * DIM,
                         wlo + (size_t)(nt + 1) * 16 * DIM, tid);
            // 4 newest ops may stay in flight; older 4 (tile nt) must be done.
            asm volatile("s_wait_asynccnt 0x4" ::: "memory");
        } else {
            asm volatile("s_wait_asynccnt 0x0" ::: "memory");
        }
        __syncthreads();

        const __bf16* bhiS = (const __bf16*)(smem + bufOff);
        const __bf16* bloS = (const __bf16*)(smem + bufOff + TILE_BYTES);

        v8f acc = {};
        #pragma unroll
        for (int kt = 0; kt < KSTEPS; ++kt) {
            // A fragment (16x32 bf16): lane m=row, h picks K-half;
            // per lane two contiguous 8-element runs: K=[8h,8h+8) and [8h+16,8h+24)
            union { v16bf v; f32x4 q[2]; } ahi, alo, bhi, blo;
            {
                const __bf16* pa = zhiS + (size_t)(rowL + m) * DIM + kt * 32 + 8 * h;
                const __bf16* pb = zloS + (size_t)(rowL + m) * DIM + kt * 32 + 8 * h;
                ahi.q[0] = *(const f32x4*)(pa);
                ahi.q[1] = *(const f32x4*)(pa + 16);
                alo.q[0] = *(const f32x4*)(pb);
                alo.q[1] = *(const f32x4*)(pb + 16);
            }
            // B fragment (32x16 bf16): lane n=m, K runs [8h,8h+8) and [8h+16,8h+24)
            {
                const __bf16* pa = bhiS + (size_t)m * DIM + kt * 32 + 8 * h;
                const __bf16* pb = bloS + (size_t)m * DIM + kt * 32 + 8 * h;
                bhi.q[0] = *(const f32x4*)(pa);
                bhi.q[1] = *(const f32x4*)(pa + 16);
                blo.q[0] = *(const f32x4*)(pb);
                blo.q[1] = *(const f32x4*)(pb + 16);
            }
            acc = __builtin_amdgcn_wmma_f32_16x16x32_bf16(false, ahi.v, false, bhi.v,
                                                          (short)0, acc, false, false);
            acc = __builtin_amdgcn_wmma_f32_16x16x32_bf16(false, ahi.v, false, blo.v,
                                                          (short)0, acc, false, false);
            acc = __builtin_amdgcn_wmma_f32_16x16x32_bf16(false, alo.v, false, bhi.v,
                                                          (short)0, acc, false, false);
        }

        // dist = ||w||^2 - 2*score  (||z||^2 dropped); running argmin.
        const int   code = nt * 16 + m;
        const float wn   = wnS[code];
        #pragma unroll
        for (int r = 0; r < 8; ++r) {
            const float dist = wn - 2.0f * acc[r];
            if (dist < bestD[r]) { bestD[r] = dist; bestI[r] = code; }  // first-min kept
        }

        __syncthreads();   // all waves done reading this buffer before reuse
    }

    // ---- argmin across the 16-lane column groups (tie -> lower index) ----
    #pragma unroll
    for (int r = 0; r < 8; ++r) {
        float d0 = bestD[r];
        int   i0 = bestI[r];
        #pragma unroll
        for (int s = 8; s >= 1; s >>= 1) {
            const float od = __shfl_xor(d0, s, 32);
            const int   oi = __shfl_xor(i0, s, 32);
            if (od < d0 || (od == d0 && oi < i0)) { d0 = od; i0 = oi; }
        }
        if (m == 0) idxS[rowL + r + 8 * h] = i0;   // h=0 -> row r, h=1 -> row r+8
    }
    __syncthreads();

    // ---- gather codewords (L2-resident), write z_q_st, accumulate mse ----
    float lsum = 0.f;
    for (int rr = 0; rr < 16; ++rr) {
        const int lrow = rowL + rr;
        const int grow = blockRow + lrow;
        const int ci   = idxS[lrow];
        const float* wr = w   + (size_t)ci * DIM;
        const float* zr = z   + (size_t)grow * DIM;
        float*       orow = out + (size_t)grow * DIM;
        for (int d = lane; d < DIM; d += 32) {
            const float zq   = wr[d];
            const float zv   = zr[d];
            const float diff = zq - zv;
            orow[d] = zv + diff;       // straight-through: z + (z_q - z)
            lsum += diff * diff;
        }
    }
    #pragma unroll
    for (int s = 16; s >= 1; s >>= 1) lsum += __shfl_xor(lsum, s, 32);
    if (lane == 0) redS[wave] = lsum;
    __syncthreads();
    if (tid == 0) {
        float t = 0.f;
        #pragma unroll
        for (int i = 0; i < 8; ++i) t += redS[i];
        partials[blockIdx.x] = t;
    }
    // indices (as float) -> d_out[B*D + 1 + row]
    if (tid < ROWS_PER_BLOCK)
        out[(size_t)B_ROWS * DIM + 1 + blockRow + tid] = (float)idxS[tid];
}

// ---------------------------------------------------------------------------
// Finalize: deterministic tree reduction of 512 block partials -> vq_loss.
// ---------------------------------------------------------------------------
__global__ __launch_bounds__(512)
void vq_finalize(const float* __restrict__ partials, float* __restrict__ out)
{
    __shared__ float s[512];
    const int t = threadIdx.x;
    s[t] = partials[t];
    __syncthreads();
    #pragma unroll
    for (int stride = 256; stride >= 1; stride >>= 1) {
        if (t < stride) s[t] += s[t + stride];
        __syncthreads();
    }
    if (t == 0) {
        const float mse = s[0] / (float)((size_t)B_ROWS * DIM);
        out[(size_t)B_ROWS * DIM] = mse * 1.25f;   // mse + 0.25*mse
    }
}

// ---------------------------------------------------------------------------
extern "C" void kernel_launch(void* const* d_in, const int* in_sizes, int n_in,
                              void* d_out, int out_size, void* d_ws, size_t ws_size,
                              hipStream_t stream)
{
    (void)in_sizes; (void)n_in; (void)out_size; (void)ws_size;
    const float* z = (const float*)d_in[0];   // [B, D] fp32
    const float* w = (const float*)d_in[1];   // [K, D] fp32
    float* out = (float*)d_out;               // [B*D] z_q_st | [1] loss | [B] indices

    char* ws = (char*)d_ws;
    __bf16* whi     = (__bf16*)(ws + 0);            // K*D*2 = 512 KB
    __bf16* wlo     = (__bf16*)(ws + (512 << 10));  // 512 KB
    float*  wnorm   = (float*)(ws + (1024 << 10));  // 4 KB
    float*  partials= (float*)(ws + (1028 << 10));  // 2 KB

    vq_prep<<<K_CODES, DIM, 0, stream>>>(w, whi, wlo, wnorm);
    vq_main<<<B_ROWS / ROWS_PER_BLOCK, 256, SMEM_BYTES, stream>>>(
        z, w, whi, wlo, wnorm, out, partials);
    vq_finalize<<<1, 512, 0, stream>>>(partials, out);
}
